// GroupedMultiQueryAttention_85770496901173
// MI455X (gfx1250) — matrix operand order repaired and checked by reading, not compiled, not run on hardware
//
#include <hip/hip_runtime.h>
#include <hip/hip_bf16.h>

// MHA forward: B=2, S=2048, D=1024, H=16, Dh=64 on gfx1250 (wave32, WMMA).
// All GEMMs (QKV proj, QK^T, PV, out proj) via v_wmma_f32_16x16x32_f16.
// Workspace: 20 Mi f16 elements = 40 MiB.

typedef _Float16 f16;
typedef __attribute__((ext_vector_type(4)))  _Float16 v4h;
typedef __attribute__((ext_vector_type(16))) _Float16 v16h;
typedef __attribute__((ext_vector_type(8)))  float    v8f;

#define D_MODEL 1024
#define SEQ     2048
#define NTOK    4096   // B*S
#define NHEAD   16
#define HDIM    64

// ---- fragment loaders (contiguous K order used by both A and B, wave32) ----

// A-matrix 16x32 (MxK): lane L holds row m=L%16; kb=L/16 selects the 16-wide K
// half: half i -> K = kb*16+i
__device__ __forceinline__ v16h load_frag_a(const f16* __restrict__ rowbase, int kb) {
    v4h c0 = *(const v4h*)(rowbase + kb * 16);
    v4h c1 = *(const v4h*)(rowbase + kb * 16 + 4);
    v4h c2 = *(const v4h*)(rowbase + kb * 16 + 8);
    v4h c3 = *(const v4h*)(rowbase + kb * 16 + 12);
    v16h a;
#pragma unroll
    for (int i = 0; i < 4; ++i) {
        a[i] = c0[i]; a[i + 4] = c1[i]; a[i + 8] = c2[i]; a[i + 12] = c3[i];
    }
    return a;
}

// B-matrix 32x16 (KxN): lane L holds col n=L%16, g=L/16; half i -> K = g*16+i.
// Caller passes pointer to 16 consecutive halves (K-contiguous storage).
__device__ __forceinline__ v16h load_frag_b16(const f16* __restrict__ base) {
    v4h c0 = *(const v4h*)(base);
    v4h c1 = *(const v4h*)(base + 4);
    v4h c2 = *(const v4h*)(base + 8);
    v4h c3 = *(const v4h*)(base + 12);
    v16h b;
#pragma unroll
    for (int i = 0; i < 4; ++i) {
        b[i] = c0[i]; b[i + 4] = c1[i]; b[i + 8] = c2[i]; b[i + 12] = c3[i];
    }
    return b;
}

// ---- conversion kernels ----

__global__ void cvt_f16_kernel(const float* __restrict__ src, f16* __restrict__ dst, int n) {
    int i = blockIdx.x * 256 + threadIdx.x;
    if (i < n) dst[i] = (f16)src[i];
}

// All four weights W (KxN fp32) -> W^T (NxK f16) in one launch (blockIdx.z picks W).
__global__ void cvt_transpose4_kernel(
    const float* __restrict__ Wq, const float* __restrict__ Wk,
    const float* __restrict__ Wv, const float* __restrict__ Wo,
    f16* __restrict__ WqT, f16* __restrict__ WkT,
    f16* __restrict__ WvT, f16* __restrict__ WoT) {
    int z = blockIdx.z;
    const float* src = (z == 0) ? Wq : (z == 1) ? Wk : (z == 2) ? Wv : Wo;
    f16* dst = (z == 0) ? WqT : (z == 1) ? WkT : (z == 2) ? WvT : WoT;
    int i = blockIdx.x * 256 + threadIdx.x;       // 0 .. 1M-1
    int k = i >> 10, n = i & 1023;
    dst[n * D_MODEL + k] = (f16)src[i];
}

// ---- tiled GEMM core: C[4096x1024] = A @ B (+bias, +optional rotary) ----
// BM=128 BN=128 BK=32; 256 threads = 8 waves (4 M x 2 N groups); each wave
// computes 32x64 = 2x4 WMMA fragments -> 8 v_wmma per K-step.
// Register-staged global loads + double-buffered LDS (1 barrier / K-step).

template <int OUT_F32>
__device__ __forceinline__ void gemm_core(
    const f16* __restrict__ A, const f16* __restrict__ BT,
    const float* __restrict__ bias, const float* __restrict__ rotary,
    f16* __restrict__ Cf16, float* __restrict__ Cf32) {
    __shared__ __attribute__((aligned(16))) f16 As[2][128 * 40];
    __shared__ __attribute__((aligned(16))) f16 Bs[2][128 * 40];

    const int t = threadIdx.x;
    const int lane = t & 31, wave = t >> 5;
    const int wm = wave & 3, wn = wave >> 2;
    const int mBase = blockIdx.y * 128;
    const int nBase = blockIdx.x * 128;
    const int mfrag = lane & 15, g = lane >> 4;

    // staging geometry: each thread moves 16 halves of A and 16 of B per tile
    const int r0 = (t * 8) >> 5;          // 0..63
    const int c0 = (t * 8) & 31;          // 0,8,16,24
    const f16* aSrc = A + (size_t)mBase * D_MODEL;
    const f16* bSrc = BT + (size_t)nBase * D_MODEL;

    v4h aR[4], bR[4];
    auto loadTile = [&](int kBase) {
        const f16* pa0 = aSrc + (size_t)r0 * D_MODEL + kBase + c0;
        const f16* pa1 = aSrc + (size_t)(r0 + 64) * D_MODEL + kBase + c0;
        aR[0] = *(const v4h*)pa0;       aR[1] = *(const v4h*)(pa0 + 4);
        aR[2] = *(const v4h*)pa1;       aR[3] = *(const v4h*)(pa1 + 4);
        const f16* pb0 = bSrc + (size_t)r0 * D_MODEL + kBase + c0;
        const f16* pb1 = bSrc + (size_t)(r0 + 64) * D_MODEL + kBase + c0;
        bR[0] = *(const v4h*)pb0;       bR[1] = *(const v4h*)(pb0 + 4);
        bR[2] = *(const v4h*)pb1;       bR[3] = *(const v4h*)(pb1 + 4);
    };
    auto storeTile = [&](int buf) {
        *(v4h*)&As[buf][r0 * 40 + c0]            = aR[0];
        *(v4h*)&As[buf][r0 * 40 + c0 + 4]        = aR[1];
        *(v4h*)&As[buf][(r0 + 64) * 40 + c0]     = aR[2];
        *(v4h*)&As[buf][(r0 + 64) * 40 + c0 + 4] = aR[3];
        *(v4h*)&Bs[buf][r0 * 40 + c0]            = bR[0];
        *(v4h*)&Bs[buf][r0 * 40 + c0 + 4]        = bR[1];
        *(v4h*)&Bs[buf][(r0 + 64) * 40 + c0]     = bR[2];
        *(v4h*)&Bs[buf][(r0 + 64) * 40 + c0 + 4] = bR[3];
    };

    v8f acc[2][4] = {};
    int buf = 0;
    loadTile(0);
    storeTile(buf);
    __syncthreads();

    const int KT = D_MODEL / 32;  // 32 K-steps
    for (int kt = 0; kt < KT; ++kt) {
        if (kt + 1 < KT) loadTile((kt + 1) * 32);
        if (kt + 2 < KT) {  // global_prefetch_b8 two steps ahead
            __builtin_prefetch(aSrc + (size_t)r0 * D_MODEL + (kt + 2) * 32 + c0, 0, 1);
            __builtin_prefetch(bSrc + (size_t)r0 * D_MODEL + (kt + 2) * 32 + c0, 0, 1);
        }

        v16h af[2], bf[4];
#pragma unroll
        for (int i = 0; i < 2; ++i)
            af[i] = load_frag_a(&As[buf][(wm * 32 + i * 16 + mfrag) * 40], g);
#pragma unroll
        for (int j = 0; j < 4; ++j)
            bf[j] = load_frag_b16(&Bs[buf][(wn * 64 + j * 16 + mfrag) * 40 + g * 16]);
#pragma unroll
        for (int i = 0; i < 2; ++i)
#pragma unroll
            for (int j = 0; j < 4; ++j)
                acc[i][j] = __builtin_amdgcn_wmma_f32_16x16x32_f16(
                    false, af[i], false, bf[j], (short)0, acc[i][j], false, false);

        if (kt + 1 < KT) storeTile(buf ^ 1);  // writes the other buffer: no race
        __syncthreads();
        buf ^= 1;
    }

    // epilogue: C layout -> lane col n = L%16, row = r + 8*(L/16)
#pragma unroll
    for (int i = 0; i < 2; ++i) {
#pragma unroll
        for (int j = 0; j < 4; ++j) {
            const int gn = nBase + wn * 64 + j * 16 + mfrag;
            float badd = bias[gn];
            if (rotary) badd += rotary[gn & (HDIM - 1)];  // uniform branch
#pragma unroll
            for (int r = 0; r < 8; ++r) {
                const int gm = mBase + wm * 32 + i * 16 + r + 8 * g;
                const float val = acc[i][j][r] + badd;
                if (OUT_F32) Cf32[(size_t)gm * D_MODEL + gn] = val;
                else         Cf16[(size_t)gm * D_MODEL + gn] = (f16)val;
            }
        }
    }
}

// Fused QKV projection: blockIdx.z in {0,1,2} selects Q (with rotary), K, V.
__global__ __launch_bounds__(256) void qkv_kernel(
    const f16* __restrict__ xh,
    const f16* __restrict__ WqT, const f16* __restrict__ WkT, const f16* __restrict__ WvT,
    const float* __restrict__ bq, const float* __restrict__ bk, const float* __restrict__ bv,
    const float* __restrict__ rotary,
    f16* __restrict__ Qh, f16* __restrict__ Kh, f16* __restrict__ Vh) {
    const int z = blockIdx.z;
    const f16* BT    = (z == 0) ? WqT : (z == 1) ? WkT : WvT;
    const float* bia = (z == 0) ? bq  : (z == 1) ? bk  : bv;
    const float* rot = (z == 0) ? rotary : nullptr;
    f16* out         = (z == 0) ? Qh  : (z == 1) ? Kh  : Vh;
    gemm_core<0>(xh, BT, bia, rot, out, nullptr);
}

// Output projection: fp32 result straight into d_out.
__global__ __launch_bounds__(256) void proj_kernel(
    const f16* __restrict__ Oh, const f16* __restrict__ WoT,
    const float* __restrict__ bo, float* __restrict__ out) {
    gemm_core<1>(Oh, WoT, bo, nullptr, nullptr, out);
}

// ---- flash attention: O = softmax(Q K^T / 8) V per (b,h) ----
// Block = 128 Q rows of one (b,h); 8 waves x 16 rows; kv streamed in blocks of 32.

__global__ __launch_bounds__(256) void attn_kernel(
    const f16* __restrict__ Q, const f16* __restrict__ K,
    const f16* __restrict__ V, f16* __restrict__ O) {
    __shared__ __attribute__((aligned(16))) f16 Ks[32 * 72];      // [kv][d]
    __shared__ __attribute__((aligned(16))) f16 VTs[64 * 40];     // [d][kv]
    __shared__ __attribute__((aligned(16))) f16 Ps[8][16 * 40];   // per-wave P [m][kv]

    const int t = threadIdx.x, lane = t & 31, wave = t >> 5;
    const int bh = blockIdx.y, b = bh >> 4, h = bh & 15;
    const int qBase = blockIdx.x * 128;
    const int mfrag = lane & 15, g = lane >> 4;

    // preload this wave's Q fragments (16 rows x 64 = 2 A-frags); rotary folded in already
    const int qrow = qBase + wave * 16 + mfrag;
    const f16* qptr = Q + (size_t)(b * SEQ + qrow) * D_MODEL + h * HDIM;
    v16h qf[2];
#pragma unroll
    for (int kc = 0; kc < 2; ++kc) qf[kc] = load_frag_a(qptr + kc * 32, g);

    float mrow[8], lrow[8];
#pragma unroll
    for (int r = 0; r < 8; ++r) { mrow[r] = -3.0e38f; lrow[r] = 0.0f; }
    v8f oacc[4] = {};

    const int ldrow = (t * 8) >> 6, ldcol = (t * 8) & 63;  // K/V staging geometry

    for (int j = 0; j < SEQ / 32; ++j) {
        // stage K block [32][64] and V^T block [64][32] in LDS
        {
            const size_t tok = (size_t)(b * SEQ + j * 32 + ldrow) * D_MODEL + h * HDIM + ldcol;
            v4h k0 = *(const v4h*)(K + tok);
            v4h k1 = *(const v4h*)(K + tok + 4);
            *(v4h*)(&Ks[ldrow * 72 + ldcol])     = k0;
            *(v4h*)(&Ks[ldrow * 72 + ldcol + 4]) = k1;
            v4h v0 = *(const v4h*)(V + tok);
            v4h v1 = *(const v4h*)(V + tok + 4);
#pragma unroll
            for (int ii = 0; ii < 4; ++ii) {
                VTs[(ldcol + ii) * 40 + ldrow]     = v0[ii];
                VTs[(ldcol + 4 + ii) * 40 + ldrow] = v1[ii];
            }
            if (j + 1 < SEQ / 32) {  // prefetch next kv block
                const size_t tokn = tok + 32 * D_MODEL;
                __builtin_prefetch(K + tokn, 0, 1);
                __builtin_prefetch(V + tokn, 0, 1);
            }
        }
        __syncthreads();

        // scores S = Q K^T : two 16x16 fragments covering 32 kv columns
        v8f s[2] = {};
#pragma unroll
        for (int nf = 0; nf < 2; ++nf) {
#pragma unroll
            for (int kc = 0; kc < 2; ++kc) {
                v16h bk = load_frag_b16(&Ks[(nf * 16 + mfrag) * 72 + kc * 32 + g * 16]);
                s[nf] = __builtin_amdgcn_wmma_f32_16x16x32_f16(
                    false, qf[kc], false, bk, (short)0, s[nf], false, false);
            }
        }

        // online softmax; row m = r + 8*g lives in this lane's 16-lane half
        const float scale = 0.125f;  // 1/sqrt(64)
#pragma unroll
        for (int r = 0; r < 8; ++r) {
            float s0 = s[0][r] * scale, s1 = s[1][r] * scale;
            float rm = fmaxf(s0, s1);
#pragma unroll
            for (int msk = 1; msk <= 8; msk <<= 1) rm = fmaxf(rm, __shfl_xor(rm, msk, 32));
            float mn = fmaxf(mrow[r], rm);
            float corr = __expf(mrow[r] - mn);
            float p0 = __expf(s0 - mn);
            float p1 = __expf(s1 - mn);
            float rs = p0 + p1;
#pragma unroll
            for (int msk = 1; msk <= 8; msk <<= 1) rs += __shfl_xor(rs, msk, 32);
            lrow[r] = lrow[r] * corr + rs;
            mrow[r] = mn;
#pragma unroll
            for (int cf = 0; cf < 4; ++cf) oacc[cf][r] *= corr;
            // scatter P (C-layout) into per-wave LDS tile in plain [m][kv] order
            Ps[wave][(r + 8 * g) * 40 + mfrag]      = (f16)p0;
            Ps[wave][(r + 8 * g) * 40 + 16 + mfrag] = (f16)p1;
        }

        // O += P V : P read back as A-fragment (same-wave LDS RAW; DS ops are
        // in-order within a wave, no barrier needed for the per-wave tile)
        v16h ap = load_frag_a(&Ps[wave][mfrag * 40], g);
#pragma unroll
        for (int cf = 0; cf < 4; ++cf) {
            v16h bv = load_frag_b16(&VTs[(cf * 16 + mfrag) * 40 + g * 16]);
            oacc[cf] = __builtin_amdgcn_wmma_f32_16x16x32_f16(
                false, ap, false, bv, (short)0, oacc[cf], false, false);
        }
        __syncthreads();
    }

    // normalize and write O token-major (B*S, H*Dh) for the output projection
#pragma unroll
    for (int cf = 0; cf < 4; ++cf) {
#pragma unroll
        for (int r = 0; r < 8; ++r) {
            const int row = qBase + wave * 16 + r + 8 * g;
            const float val = oacc[cf][r] / lrow[r];
            O[(size_t)(b * SEQ + row) * D_MODEL + h * HDIM + cf * 16 + mfrag] = (f16)val;
        }
    }
}

extern "C" void kernel_launch(void* const* d_in, const int* in_sizes, int n_in,
                              void* d_out, int out_size, void* d_ws, size_t ws_size,
                              hipStream_t stream) {
    (void)in_sizes; (void)n_in; (void)out_size; (void)ws_size;
    const float* x      = (const float*)d_in[0];
    const float* Wq     = (const float*)d_in[1];
    const float* bq     = (const float*)d_in[2];
    const float* Wk     = (const float*)d_in[3];
    const float* bk     = (const float*)d_in[4];
    const float* Wv     = (const float*)d_in[5];
    const float* bv     = (const float*)d_in[6];
    const float* Wo     = (const float*)d_in[7];
    const float* bo     = (const float*)d_in[8];
    const float* rotary = (const float*)d_in[9];

    f16* ws = (f16*)d_ws;
    const size_t NX = (size_t)NTOK * D_MODEL;     // 4 Mi elems
    const size_t NW = (size_t)D_MODEL * D_MODEL;  // 1 Mi elems
    f16* xh  = ws;                 // reused as attention output O
    f16* WqT = xh + NX;
    f16* WkT = WqT + NW;
    f16* WvT = WkT + NW;
    f16* WoT = WvT + NW;
    f16* Qh  = WoT + NW;
    f16* Kh  = Qh + NX;
    f16* Vh  = Kh + NX;
    f16* Oh  = xh;                 // x dead after QKV GEMMs

    cvt_f16_kernel<<<(int)(NX / 256), 256, 0, stream>>>(x, xh, (int)NX);
    cvt_transpose4_kernel<<<dim3((unsigned)(NW / 256), 1, 4), 256, 0, stream>>>(
        Wq, Wk, Wv, Wo, WqT, WkT, WvT, WoT);

    dim3 qkvGrid(D_MODEL / 128, NTOK / 128, 3);   // (8, 32, 3)
    qkv_kernel<<<qkvGrid, 256, 0, stream>>>(xh, WqT, WkT, WvT, bq, bk, bv, rotary,
                                            Qh, Kh, Vh);

    attn_kernel<<<dim3(SEQ / 128, 2 * NHEAD), 256, 0, stream>>>(Qh, Kh, Vh, Oh);

    dim3 projGrid(D_MODEL / 128, NTOK / 128);     // (8, 32)
    proj_kernel<<<projGrid, 256, 0, stream>>>(Oh, WoT, bo, (float*)d_out);
}
